// modelLSTMVent_72533407695099
// MI455X (gfx1250) — matrix operand order repaired and checked by reading, not compiled, not run on hardware
//
#include <hip/hip_runtime.h>
#include <math.h>

#define BATCH 8192
#define NF    12
#define NT    20
#define NH    32
#define G4H   128   // 4*H

typedef __attribute__((ext_vector_type(16))) _Float16 v16h;
typedef __attribute__((ext_vector_type(8)))  float    v8f;

// Branch-free activations: v_exp_f32 + v_rcp_f32, no EXEC manipulation.
__device__ __forceinline__ float sigf(float x) {
    return __builtin_amdgcn_rcpf(1.0f + __expf(-x));
}
__device__ __forceinline__ float tanh_fast(float x) {
    return 2.0f * __builtin_amdgcn_rcpf(1.0f + __expf(-2.0f * x)) - 1.0f;
}

// Scheduling fence: stops LLVM from hoisting the loop-invariant LDS weight-
// fragment loads out of the time loop (which previously spilled to scratch).
__device__ __forceinline__ void sched_fence() { asm volatile("" ::: "memory"); }

// One wave = 16 batches x 1 feature.  Block = 256 threads = 8 waves, all on the
// same feature (shared weight fragments in LDS).  grid = (B/128, F).
__global__ __launch_bounds__(256, 4)
void lstm_wmma_kernel(const float* __restrict__ x,
                      const float* __restrict__ W_ih0,  // (F,128,1)
                      const float* __restrict__ W_hh0,  // (F,128,32)
                      const float* __restrict__ b_ih0,  // (F,128)
                      const float* __restrict__ b_hh0,  // (F,128)
                      const float* __restrict__ W_ih1,  // (F,128,32)
                      const float* __restrict__ W_hh1,  // (F,128,32)
                      const float* __restrict__ b_ih1,  // (F,128)
                      const float* __restrict__ b_hh1,  // (F,128)
                      const float* __restrict__ W_f1,   // (1,32)
                      const float* __restrict__ b_f1,   // (1,)
                      const float* __restrict__ W_f2,   // (1,32)
                      const float* __restrict__ b_f2,   // (1,)
                      const float* __restrict__ W_t,    // (1,F*32)
                      float* __restrict__ ws_tgt,       // (F,B) partial target dots
                      float* __restrict__ out_feats)    // (B,11)
{
    // B-fragment layout used here (contiguous k, same order as A fragments):
    //   lane L: n = tile*16 + (L&15);  half h (0..15): K = h + 16*(L>>4)
    __shared__ alignas(32) _Float16 wf_hh0[8][32][16];
    __shared__ alignas(32) _Float16 wf_ih1[8][32][16];
    __shared__ alignas(32) _Float16 wf_hh1[8][32][16];
    __shared__ float s_wih0[G4H];
    __shared__ float s_b0[G4H];
    __shared__ float s_b1[G4H];
    // Per-wave h staging buffer (row-major 16 x 32, f16) for D->A relayout.
    __shared__ alignas(16) _Float16 hstage[8][16][32];

    const int tid  = threadIdx.x;
    const int f    = blockIdx.y;
    const int wave = tid >> 5;
    const int lane = tid & 31;
    const int nlo  = lane & 15;     // column within a 16-wide tile
    const int hi   = lane >> 4;     // lane half (0 or 1)

    // ---- Stage weights into LDS as f16 B-fragments (one (tile,lane) per thread)
    {
        const int tile = tid >> 5;          // 0..7
        const int ln   = tid & 31;          // 0..31
        const int n    = tile * 16 + (ln & 15);
        const int kb   = (ln >> 4) * 16;
        const float* w0 = W_hh0 + (size_t)f * G4H * NH + (size_t)n * NH + kb;
        const float* w1 = W_ih1 + (size_t)f * G4H * NH + (size_t)n * NH + kb;
        const float* w2 = W_hh1 + (size_t)f * G4H * NH + (size_t)n * NH + kb;
#pragma unroll
        for (int k = 0; k < 16; ++k) {
            wf_hh0[tile][ln][k] = (_Float16)w0[k];
            wf_ih1[tile][ln][k] = (_Float16)w1[k];
            wf_hh1[tile][ln][k] = (_Float16)w2[k];
        }
        if (tid < G4H) {
            s_wih0[tid] = W_ih0[f * G4H + tid];
            s_b0[tid]   = b_ih0[f * G4H + tid] + b_hh0[f * G4H + tid];
            s_b1[tid]   = b_ih1[f * G4H + tid] + b_hh1[f * G4H + tid];
        }
    }
    __syncthreads();

    const int gbase = (blockIdx.x * 8 + wave) * 16;   // first batch of this wave

    // Per-row global x offsets (row r of this lane -> batch gbase + r + hi*8)
    int xoff[8];
#pragma unroll
    for (int r = 0; r < 8; ++r) xoff[r] = (gbase + r + hi * 8) * NF * NT;
    const int foff = f * NT;

    // State (f32, D-layout): tile p covers hidden cols [16p, 16p+16)
    v8f c0[2] = {}, c1[2] = {}, h0d[2] = {}, h1d[2] = {}, outd[2] = {};
    float aliveF[8];
#pragma unroll
    for (int r = 0; r < 8; ++r) aliveF[r] = 1.0f;

    // D-layout -> A-fragment via per-wave LDS staging.
    auto buildAfrag = [&](const v8f hd0, const v8f hd1) -> v16h {
#pragma unroll
        for (int r = 0; r < 8; ++r) {
            const int m = r + hi * 8;
            hstage[wave][m][nlo]      = (_Float16)hd0[r];
            hstage[wave][m][16 + nlo] = (_Float16)hd1[r];
        }
        union { v16h h; unsigned u[8]; } af;
        const int m    = lane & 15;
        const int koff = hi * 16;
#pragma unroll
        for (int v = 0; v < 8; ++v) {
            const int K = v * 2 + koff;
            af.u[v] = *reinterpret_cast<const unsigned*>(&hstage[wave][m][K]);
        }
        return af.h;
    };

    auto loadB = [&](const _Float16 (*wf)[32][16], int g) -> v16h {
        return *reinterpret_cast<const v16h*>(&wf[g][lane][0]);
    };

    const v8f zero8 = {};

    for (int t = 0; t < NT; ++t) {
        sched_fence();
        // alive &= (x[:,0,t] != 0);  gather x[:,f,t] for the rank-1 input term
        float xv[8];
#pragma unroll
        for (int r = 0; r < 8; ++r) {
            const float x0 = x[xoff[r] + t];          // feature 0
            aliveF[r] = (x0 != 0.0f) ? aliveF[r] : 0.0f;
            xv[r] = x[xoff[r] + foff + t];
        }

        // Fragments of current h0, h1 (sequential reuse of the staging buffer)
        v16h a_h0 = buildAfrag(h0d[0], h0d[1]);
        v16h a_h1 = buildAfrag(h1d[0], h1d[1]);

        // ---- Layer 0: z0 = h0 @ W_hh0 + b0 + x*W_ih0, one column-half at a time
#pragma unroll
        for (int p = 0; p < 2; ++p) {
            sched_fence();   // keep this group's B fragments in LDS until here
            v8f zi = __builtin_amdgcn_wmma_f32_16x16x32_f16(
                false, a_h0, false, loadB(wf_hh0, 0 + p), (short)0, zero8, false, false);
            v8f zf = __builtin_amdgcn_wmma_f32_16x16x32_f16(
                false, a_h0, false, loadB(wf_hh0, 2 + p), (short)0, zero8, false, false);
            v8f zg = __builtin_amdgcn_wmma_f32_16x16x32_f16(
                false, a_h0, false, loadB(wf_hh0, 4 + p), (short)0, zero8, false, false);
            v8f zo = __builtin_amdgcn_wmma_f32_16x16x32_f16(
                false, a_h0, false, loadB(wf_hh0, 6 + p), (short)0, zero8, false, false);
            const float bi = s_b0[(0 + p) * 16 + nlo], wii = s_wih0[(0 + p) * 16 + nlo];
            const float bf = s_b0[(2 + p) * 16 + nlo], wif = s_wih0[(2 + p) * 16 + nlo];
            const float bg = s_b0[(4 + p) * 16 + nlo], wig = s_wih0[(4 + p) * 16 + nlo];
            const float bo = s_b0[(6 + p) * 16 + nlo], wio = s_wih0[(6 + p) * 16 + nlo];
#pragma unroll
            for (int e = 0; e < 8; ++e) {
                const float ig = sigf(zi[e] + bi + xv[e] * wii);
                const float fg = sigf(zf[e] + bf + xv[e] * wif);
                const float gg = tanh_fast(zg[e] + bg + xv[e] * wig);
                const float og = sigf(zo[e] + bo + xv[e] * wio);
                const float cn = fg * c0[p][e] + ig * gg;
                const float hn = og * tanh_fast(cn);
                const float a  = aliveF[e];
                c0[p][e]  = a * (cn - c0[p][e]) + c0[p][e];
                h0d[p][e] = a * (hn - h0d[p][e]) + h0d[p][e];
            }
        }

        // ---- Layer 1: z1 = h0_new @ W_ih1 + h1 @ W_hh1 + b1
        v16h a_h0n = buildAfrag(h0d[0], h0d[1]);
#pragma unroll
        for (int p = 0; p < 2; ++p) {
            sched_fence();   // keep this group's B fragments in LDS until here
            v8f zi = __builtin_amdgcn_wmma_f32_16x16x32_f16(
                false, a_h1, false, loadB(wf_hh1, 0 + p), (short)0,
                __builtin_amdgcn_wmma_f32_16x16x32_f16(
                    false, a_h0n, false, loadB(wf_ih1, 0 + p), (short)0, zero8, false, false),
                false, false);
            v8f zf = __builtin_amdgcn_wmma_f32_16x16x32_f16(
                false, a_h1, false, loadB(wf_hh1, 2 + p), (short)0,
                __builtin_amdgcn_wmma_f32_16x16x32_f16(
                    false, a_h0n, false, loadB(wf_ih1, 2 + p), (short)0, zero8, false, false),
                false, false);
            v8f zg = __builtin_amdgcn_wmma_f32_16x16x32_f16(
                false, a_h1, false, loadB(wf_hh1, 4 + p), (short)0,
                __builtin_amdgcn_wmma_f32_16x16x32_f16(
                    false, a_h0n, false, loadB(wf_ih1, 4 + p), (short)0, zero8, false, false),
                false, false);
            v8f zo = __builtin_amdgcn_wmma_f32_16x16x32_f16(
                false, a_h1, false, loadB(wf_hh1, 6 + p), (short)0,
                __builtin_amdgcn_wmma_f32_16x16x32_f16(
                    false, a_h0n, false, loadB(wf_ih1, 6 + p), (short)0, zero8, false, false),
                false, false);
            const float bi = s_b1[(0 + p) * 16 + nlo];
            const float bf = s_b1[(2 + p) * 16 + nlo];
            const float bg = s_b1[(4 + p) * 16 + nlo];
            const float bo = s_b1[(6 + p) * 16 + nlo];
#pragma unroll
            for (int e = 0; e < 8; ++e) {
                const float ig = sigf(zi[e] + bi);
                const float fg = sigf(zf[e] + bf);
                const float gg = tanh_fast(zg[e] + bg);
                const float og = sigf(zo[e] + bo);
                const float cn = fg * c1[p][e] + ig * gg;
                const float hn = og * tanh_fast(cn);
                const float a  = aliveF[e];
                c1[p][e]   = a * (cn - c1[p][e]) + c1[p][e];
                h1d[p][e]  = a * (hn - h1d[p][e]) + h1d[p][e];
                outd[p][e] = a * (hn - outd[p][e]) + outd[p][e];
            }
        }
    }

    // ---- Heads.  Dot over hidden dim = reduce over the 16 lanes of each half.
    const float wtl = W_t[f * NH + nlo];
    const float wth = W_t[f * NH + 16 + nlo];
    float wfl = 0.0f, wfh = 0.0f, fbias = 0.0f;
    int  col = -1;
    if (f == 1)      { wfl = W_f1[nlo]; wfh = W_f1[16 + nlo]; fbias = b_f1[0]; col = 0; }
    else if (f >= 2) { wfl = W_f2[nlo]; wfh = W_f2[16 + nlo]; fbias = b_f2[0]; col = f - 1; }

#pragma unroll
    for (int r = 0; r < 8; ++r) {
        float vt = outd[0][r] * wtl + outd[1][r] * wth;
        float vf = outd[0][r] * wfl + outd[1][r] * wfh;
#pragma unroll
        for (int m = 1; m < 16; m <<= 1) {
            vt += __shfl_xor(vt, m, 32);
            vf += __shfl_xor(vf, m, 32);
        }
        if (nlo == 0) {
            const int gb = gbase + r + hi * 8;
            ws_tgt[(size_t)f * BATCH + gb] = vt;
            if (col >= 0) out_feats[(size_t)gb * 11 + col] = vf + fbias;
        }
    }
}

__global__ __launch_bounds__(256)
void head_reduce_kernel(const float* __restrict__ ws_tgt,
                        const float* __restrict__ b_t,
                        float* __restrict__ out_tgt)
{
    const int b = blockIdx.x * blockDim.x + threadIdx.x;
    if (b < BATCH) {
        float s = b_t[0];
#pragma unroll
        for (int f = 0; f < NF; ++f) s += ws_tgt[(size_t)f * BATCH + b];
        out_tgt[b] = s;
    }
}

extern "C" void kernel_launch(void* const* d_in, const int* in_sizes, int n_in,
                              void* d_out, int out_size, void* d_ws, size_t ws_size,
                              hipStream_t stream) {
    (void)in_sizes; (void)n_in; (void)out_size; (void)ws_size;
    const float* x     = (const float*)d_in[0];
    const float* W_ih0 = (const float*)d_in[1];
    const float* W_hh0 = (const float*)d_in[2];
    const float* b_ih0 = (const float*)d_in[3];
    const float* b_hh0 = (const float*)d_in[4];
    const float* W_ih1 = (const float*)d_in[5];
    const float* W_hh1 = (const float*)d_in[6];
    const float* b_ih1 = (const float*)d_in[7];
    const float* b_hh1 = (const float*)d_in[8];
    const float* W_f1  = (const float*)d_in[9];
    const float* b_f1  = (const float*)d_in[10];
    const float* W_f2  = (const float*)d_in[11];
    const float* b_f2  = (const float*)d_in[12];
    const float* W_t   = (const float*)d_in[13];
    const float* b_t   = (const float*)d_in[14];

    float* out_tgt   = (float*)d_out;            // (B,)
    float* out_feats = (float*)d_out + BATCH;    // (B,11)
    float* ws_tgt    = (float*)d_ws;             // (F,B) partial target dots

    dim3 grid(BATCH / 128, NF);
    lstm_wmma_kernel<<<grid, 256, 0, stream>>>(
        x, W_ih0, W_hh0, b_ih0, b_hh0, W_ih1, W_hh1, b_ih1, b_hh1,
        W_f1, b_f1, W_f2, b_f2, W_t, ws_tgt, out_feats);

    head_reduce_kernel<<<(BATCH + 255) / 256, 256, 0, stream>>>(ws_tgt, b_t, out_tgt);
}
